// ReLULSTM_45638322487783
// MI455X (gfx1250) — matrix operand order repaired and checked by reading, not compiled, not run on hardware
//
#include <hip/hip_runtime.h>
#include <cstdint>

// Problem sizes (fixed by the reference)
#define S_LEN 2048
#define BATCH 128
#define HID   256
#define INP   256
#define GATES 1024   // 4*HID
#define NWG   16     // persistent workgroups; each owns 16 hidden units (64 gate cols)
#define KB0   16     // layer-0 K blocks of 32 (K=512: kb 0..7 from x, 8..15 from h)
#define KB1   8      // layer-1 K blocks of 32 (K=256, weights pre-summed)

typedef __attribute__((ext_vector_type(16))) __bf16         v16bf;
typedef __attribute__((ext_vector_type(8)))  float          v8f;
typedef __attribute__((ext_vector_type(16))) unsigned short v16us;

union V16 {
  v16us v;
  uint4 q[2];
  unsigned short s[16];
};

__device__ __forceinline__ unsigned short f2bf(float f) {
  unsigned u = __float_as_uint(f);
  u += 0x7FFFu + ((u >> 16) & 1u);   // round-to-nearest-even
  return (unsigned short)(u >> 16);
}
__device__ __forceinline__ float sigf(float x) {
  return 1.0f / (1.0f + __expf(-x));
}

// -------------------------------------------------------------------------
// Prep: build bf16 weight images pre-swizzled into the wave32 WMMA B-layout
// (B 16-bit 32x16: lane l -> column l&15; halves i=0..15 -> K = (l>>4)*16+i, this code's own K order, same as A),
// pre-sum layer-1 weights and both bias pairs, bf16-convert h0, zero sync.
// -------------------------------------------------------------------------
__global__ void relulstm_prep(
    const float* __restrict__ w_ih, const float* __restrict__ b_ih,
    const float* __restrict__ w_hh, const float* __restrict__ b_hh,
    const float* __restrict__ h0,
    unsigned short* __restrict__ B0, unsigned short* __restrict__ B1,
    float* __restrict__ biasc, unsigned short* __restrict__ hA,
    unsigned* __restrict__ syncw)
{
  const int SZ0 = NWG * KB0 * 4 * 32 * 16;  // 524288
  const int SZ1 = NWG * KB1 * 4 * 32 * 16;  // 262144
  int id = blockIdx.x * blockDim.x + threadIdx.x;

  if (id < SZ0) {                    // layer 0 fused [K=512 x 1024] slice images
    int i  = id & 15;
    int l  = (id >> 4) & 31;
    int nb = (id >> 9) & 3;
    int kb = (id >> 11) & 15;
    int g  = id >> 15;
    int col = nb * HID + g * 16 + (l & 15);          // gate column (row of W)
    int k   = kb * 32 + ((l >> 4) * 16) + i;         // WMMA B swizzle
    float v = (k < INP) ? w_ih[col * INP + k]        // x contribution
                        : w_hh[col * HID + (k - INP)]; // h contribution
    B0[id] = f2bf(v);
    return;
  }
  id -= SZ0;
  if (id < SZ1) {                    // layer 1 pre-summed [K=256 x 1024]
    int i  = id & 15;
    int l  = (id >> 4) & 31;
    int nb = (id >> 9) & 3;
    int kb = (id >> 11) & 7;
    int g  = id >> 14;
    int col = nb * HID + g * 16 + (l & 15);
    int k   = kb * 32 + ((l >> 4) * 16) + i;
    float v = w_ih[GATES * INP + col * INP + k] + w_hh[GATES * HID + col * HID + k];
    B1[id] = f2bf(v);
    return;
  }
  id -= SZ1;
  if (id < 2 * GATES) {              // combined biases, both layers
    biasc[id] = b_ih[id] + b_hh[id];
    return;
  }
  id -= 2 * GATES;
  if (id < BATCH * HID) {            // h state init = h0[0], bf16
    hA[id] = f2bf(h0[id]);
    return;
  }
  id -= BATCH * HID;
  if (id < 16) syncw[id] = 0u;       // grid-sync counters
}

// -------------------------------------------------------------------------
// Device-wide sync (16 persistent WGs) with s_sleep backoff.
// -------------------------------------------------------------------------
__device__ __forceinline__ void grid_sync(unsigned* cnt, unsigned* rel, unsigned target)
{
  __threadfence();          // release: make h-buffer stores visible at device scope
  __syncthreads();
  if (threadIdx.x == 0) {
    unsigned arrived = atomicAdd(cnt, 1u) + 1u;
    if (arrived == target * NWG) {
      atomicExch(rel, target);
    } else {
      while (atomicAdd(rel, 0u) < target) {
        __builtin_amdgcn_s_sleep(2);
      }
    }
  }
  __syncthreads();
  __threadfence();          // acquire
}

// -------------------------------------------------------------------------
// Persistent recurrence kernel: 16 WGs x 256 threads (8 wave32).
// WG g owns hidden units [g*16, g*16+16): weight slices resident in LDS.
// Wave w owns batch rows [16w,16w+16): i/f/g/o gate tiles + c-state live in
// f32 WMMA accumulator registers across all 2048 steps.
// -------------------------------------------------------------------------
__global__ void __launch_bounds__(256, 1) relulstm_persist(
    const float* __restrict__ x, const float* __restrict__ c0,
    const unsigned short* __restrict__ B0, const unsigned short* __restrict__ B1,
    const float* __restrict__ biasc,
    unsigned short* __restrict__ hA, unsigned short* __restrict__ hB,
    unsigned* __restrict__ syncw, float* __restrict__ out)
{
  extern __shared__ unsigned char smem[];
  unsigned short* w0 = (unsigned short*)smem;    // 32768 ushorts (64KB): layer-0 slice
  unsigned short* w1 = w0 + KB0 * 4 * 32 * 16;   // 16384 ushorts (32KB): layer-1 slice

  const int g    = blockIdx.x;          // 0..15
  const int tid  = threadIdx.x;
  const int lane = tid & 31;
  const int wid  = tid >> 5;            // 0..7

  // One-time cooperative fill of resident weight slices (16B vectors).
  {
    const uint4* s0 = (const uint4*)B0 + (size_t)g * (KB0 * 4 * 32 * 16 / 8);
    uint4* d0 = (uint4*)w0;
    for (int i = tid; i < KB0 * 4 * 32 * 16 / 8; i += 256) d0[i] = s0[i];
    const uint4* s1 = (const uint4*)B1 + (size_t)g * (KB1 * 4 * 32 * 16 / 8);
    uint4* d1 = (uint4*)w1;
    for (int i = tid; i < KB1 * 4 * 32 * 16 / 8; i += 256) d1[i] = s1[i];
  }
  __syncthreads();

  const int row0  = wid * 16;                      // wave's batch-row base
  const int nloc  = lane & 15;                     // column within 16-wide tile
  const int lh    = (lane >> 4) * 16;              // A-layout K-half select
  const int rowst = row0 + ((lane >= 16) ? 8 : 0); // C/D layout row base
  const int col   = g * 16 + nloc;                 // global hidden-unit index

  // Combined biases for this lane's column (registers, read once).
  const float bi0 = biasc[0 * HID + col], bf0 = biasc[1 * HID + col];
  const float bg0 = biasc[2 * HID + col], bo0 = biasc[3 * HID + col];
  const float bi1 = biasc[GATES + 0 * HID + col], bf1 = biasc[GATES + 1 * HID + col];
  const float bg1 = biasc[GATES + 2 * HID + col], bo1 = biasc[GATES + 3 * HID + col];

  // Running cell state in C/D register layout.
  v8f cst;
#pragma unroll
  for (int r = 0; r < 8; ++r) cst[r] = c0[(rowst + r) * HID + col];

  unsigned phase = 0;

  for (int t = 0; t < S_LEN; ++t) {
    // ================= cell 0: gates = [x_t ; h] @ [Wih0 ; Whh0]^T =========
    v8f acc[4];
#pragma unroll
    for (int G = 0; G < 4; ++G)
#pragma unroll
      for (int r = 0; r < 8; ++r) acc[G][r] = 0.f;

    const float*          xrow = x + ((size_t)t * BATCH + row0 + nloc) * INP;
    const unsigned short* hrow = hA + (row0 + nloc) * HID;

    // x contribution (fp32 -> bf16 on the fly), K blocks 0..7
    for (int kb = 0; kb < 8; ++kb) {
      const float* xp = xrow + kb * 32 + lh;
      float4 f0 = *(const float4*)(xp);
      float4 f1 = *(const float4*)(xp + 4);
      float4 f2 = *(const float4*)(xp + 8);
      float4 f3 = *(const float4*)(xp + 12);
      V16 a;
      a.s[0]  = f2bf(f0.x); a.s[1]  = f2bf(f0.y); a.s[2]  = f2bf(f0.z); a.s[3]  = f2bf(f0.w);
      a.s[4]  = f2bf(f1.x); a.s[5]  = f2bf(f1.y); a.s[6]  = f2bf(f1.z); a.s[7]  = f2bf(f1.w);
      a.s[8]  = f2bf(f2.x); a.s[9]  = f2bf(f2.y); a.s[10] = f2bf(f2.z); a.s[11] = f2bf(f2.w);
      a.s[12] = f2bf(f3.x); a.s[13] = f2bf(f3.y); a.s[14] = f2bf(f3.z); a.s[15] = f2bf(f3.w);
      v16bf av = __builtin_bit_cast(v16bf, a.v);
#pragma unroll
      for (int nb = 0; nb < 4; ++nb) {
        V16 b;
        const uint4* wp = (const uint4*)(w0 + ((size_t)(kb * 4 + nb) * 32 + lane) * 16);
        b.q[0] = wp[0]; b.q[1] = wp[1];
        v16bf bv = __builtin_bit_cast(v16bf, b.v);
        acc[nb] = __builtin_amdgcn_wmma_f32_16x16x32_bf16(
            false, av, false, bv, (short)0, acc[nb], false, false);
      }
    }
    // h contribution (bf16 from hA), K blocks 8..15 of the fused slice
    for (int kb = 0; kb < 8; ++kb) {
      V16 a;
      const unsigned short* hp = hrow + kb * 32 + lh;
      a.q[0] = *(const uint4*)(hp);
      a.q[1] = *(const uint4*)(hp + 8);
      v16bf av = __builtin_bit_cast(v16bf, a.v);
#pragma unroll
      for (int nb = 0; nb < 4; ++nb) {
        V16 b;
        const uint4* wp = (const uint4*)(w0 + ((size_t)((kb + 8) * 4 + nb) * 32 + lane) * 16);
        b.q[0] = wp[0]; b.q[1] = wp[1];
        v16bf bv = __builtin_bit_cast(v16bf, b.v);
        acc[nb] = __builtin_amdgcn_wmma_f32_16x16x32_bf16(
            false, av, false, bv, (short)0, acc[nb], false, false);
      }
    }

    // ReLU-LSTM cell (f32, directly on accumulator layout)
    v8f hv;
#pragma unroll
    for (int r = 0; r < 8; ++r) {
      float ig = sigf(acc[0][r] + bi0);
      float fg = sigf(acc[1][r] + bf0);
      float gg = fmaxf(acc[2][r] + bg0, 0.f);
      float og = sigf(acc[3][r] + bo0);
      float c  = fg * cst[r] + ig * gg;
      cst[r] = c;
      hv[r]  = og * fmaxf(c, 0.f);
    }
#pragma unroll
    for (int r = 0; r < 8; ++r)
      hB[(rowst + r) * HID + col] = f2bf(hv[r]);
    if (t == S_LEN - 1) {
      float* hy0 = out + (size_t)S_LEN * BATCH * HID;
      float* cy0 = hy0 + 2 * BATCH * HID;
#pragma unroll
      for (int r = 0; r < 8; ++r) {
        hy0[(rowst + r) * HID + col] = hv[r];
        cy0[(rowst + r) * HID + col] = cst[r];
      }
    }
    grid_sync(&syncw[0], &syncw[1], ++phase);

    // ================= cell 1: gates = h1 @ (Wih1+Whh1)^T ==================
#pragma unroll
    for (int G = 0; G < 4; ++G)
#pragma unroll
      for (int r = 0; r < 8; ++r) acc[G][r] = 0.f;

    const unsigned short* h1row = hB + (row0 + nloc) * HID;
    for (int kb = 0; kb < KB1; ++kb) {
      V16 a;
      const unsigned short* hp = h1row + kb * 32 + lh;
      a.q[0] = *(const uint4*)(hp);
      a.q[1] = *(const uint4*)(hp + 8);
      v16bf av = __builtin_bit_cast(v16bf, a.v);
#pragma unroll
      for (int nb = 0; nb < 4; ++nb) {
        V16 b;
        const uint4* wp = (const uint4*)(w1 + ((size_t)(kb * 4 + nb) * 32 + lane) * 16);
        b.q[0] = wp[0]; b.q[1] = wp[1];
        v16bf bv = __builtin_bit_cast(v16bf, b.v);
        acc[nb] = __builtin_amdgcn_wmma_f32_16x16x32_bf16(
            false, av, false, bv, (short)0, acc[nb], false, false);
      }
    }

#pragma unroll
    for (int r = 0; r < 8; ++r) {
      float ig = sigf(acc[0][r] + bi1);
      float fg = sigf(acc[1][r] + bf1);
      float gg = fmaxf(acc[2][r] + bg1, 0.f);
      float og = sigf(acc[3][r] + bo1);
      float c  = fg * cst[r] + ig * gg;
      cst[r] = c;
      hv[r]  = og * fmaxf(c, 0.f);
    }
#pragma unroll
    for (int r = 0; r < 8; ++r) {
      hA[(rowst + r) * HID + col] = f2bf(hv[r]);                       // next-step h
      out[((size_t)t * BATCH + rowst + r) * HID + col] = hv[r];        // per-step output
    }
    if (t == S_LEN - 1) {
      float* hy1 = out + (size_t)S_LEN * BATCH * HID + BATCH * HID;
      float* cy1 = hy1 + 2 * BATCH * HID;
#pragma unroll
      for (int r = 0; r < 8; ++r) {
        hy1[(rowst + r) * HID + col] = hv[r];
        cy1[(rowst + r) * HID + col] = cst[r];
      }
    }
    grid_sync(&syncw[0], &syncw[1], ++phase);
  }
}

// -------------------------------------------------------------------------
// Host launch
// -------------------------------------------------------------------------
extern "C" void kernel_launch(void* const* d_in, const int* in_sizes, int n_in,
                              void* d_out, int out_size, void* d_ws, size_t ws_size,
                              hipStream_t stream)
{
  const float* x    = (const float*)d_in[0];
  const float* h0   = (const float*)d_in[1];
  const float* c0   = (const float*)d_in[2];
  const float* w_ih = (const float*)d_in[3];
  const float* b_ih = (const float*)d_in[4];
  const float* w_hh = (const float*)d_in[5];
  const float* b_hh = (const float*)d_in[6];
  float* out = (float*)d_out;

  char* ws = (char*)d_ws;
  unsigned short* B0    = (unsigned short*)(ws);              // 1,048,576 B
  unsigned short* B1    = (unsigned short*)(ws + 1048576);    //   524,288 B
  float*          biasc = (float*)(ws + 1572864);             //     8,192 B
  unsigned short* hA    = (unsigned short*)(ws + 1581056);    //    65,536 B
  unsigned short* hB    = (unsigned short*)(ws + 1646592);    //    65,536 B
  unsigned*       syncw = (unsigned*)(ws + 1712128);          //        64 B

  const int smem_bytes = (KB0 * 4 * 32 * 16 + KB1 * 4 * 32 * 16) * (int)sizeof(unsigned short); // 98304
  hipFuncSetAttribute((const void*)relulstm_persist,
                      hipFuncAttributeMaxDynamicSharedMemorySize, smem_bytes);

  const int total = 524288 + 262144 + 2048 + 32768 + 16;
  relulstm_prep<<<(total + 255) / 256, 256, 0, stream>>>(
      w_ih, b_ih, w_hh, b_hh, h0, B0, B1, biasc, hA, syncw);

  relulstm_persist<<<NWG, 256, smem_bytes, stream>>>(
      x, c0, B0, B1, biasc, hA, hB, syncw, out);
}
